// TransformerBlock_11536282157174
// MI455X (gfx1250) — matrix operand order repaired and checked by reading, not compiled, not run on hardware
//
#include <hip/hip_runtime.h>
#include <hip/hip_bf16.h>

// ---------------------------------------------------------------------------
// MI455X (gfx1250) transformer block: bf16 WMMA everywhere, f32 accumulate.
// Compute-bound (~160 GFLOP vs ~6us of HBM traffic at 23.3 TB/s), so all
// GEMMs + attention run on v_wmma_f32_16x16x32_bf16. Routed MoE is gathered
// (top-2 only). Weights are converted fp32->bf16 transposed once per launch.
// GEMM tiles are staged GLOBAL->LDS with the CDNA5 async-tensor path
// (global_load_async_to_lds_b128 + s_wait_asynccnt).
// ---------------------------------------------------------------------------

typedef __attribute__((ext_vector_type(16))) __bf16 v16bf;
typedef __attribute__((ext_vector_type(8)))  __bf16 v8bf;
typedef __attribute__((ext_vector_type(8)))  float  v8f;

#define DEV static __device__ __forceinline__

constexpr int   Bb   = 2;
constexpr int   T    = 2048;
constexpr int   D    = 1024;
constexpr int   H    = 16;
constexpr int   KVH  = 4;
constexpr int   HD   = 64;
constexpr int   E    = 8;
constexpr int   TOPK = 2;
constexpr int   FFH  = 2048;
constexpr int   N    = Bb * T;            // 4096 tokens
constexpr int   QKV  = D + 2 * KVH * HD;  // 1536
constexpr float EPS  = 1e-6f;
constexpr float LOG2E = 1.4426950408889634f;

DEV __bf16 f2bf(float f) { return (__bf16)f; }

DEV v8f wmma_bf16(v16bf a, v16bf b, v8f c) {
  return __builtin_amdgcn_wmma_f32_16x16x32_bf16(false, a, false, b, (short)0, c,
                                                 false, false);
}

DEV v8f zero8f() {
  v8f z;
#pragma unroll
  for (int i = 0; i < 8; ++i) z[i] = 0.0f;
  return z;
}

DEV v16bf zero16bf() {
  v16bf z;
#pragma unroll
  for (int i = 0; i < 16; ++i) z[i] = (__bf16)0.0f;
  return z;
}

// LDS byte offset of a shared-memory pointer (AS3 pointer value == LDS addr)
typedef __attribute__((address_space(3))) const void lds_cvoid;
DEV unsigned lds_byte_addr(const void* p) {
  return (unsigned)(size_t)(lds_cvoid*)p;
}

// CDNA5 async GLOBAL -> LDS copies (tracked by ASYNCcnt)
DEV void async_b128(unsigned ldsaddr, const void* g) {
  asm volatile("global_load_async_to_lds_b128 %0, %1, off"
               :: "v"(ldsaddr), "v"((unsigned long long)(size_t)g) : "memory");
}
DEV void async_b128_o16(unsigned ldsaddr, const void* g) {
  asm volatile("global_load_async_to_lds_b128 %0, %1, off offset:16"
               :: "v"(ldsaddr), "v"((unsigned long long)(size_t)g) : "memory");
}
DEV void wait_async0() { asm volatile("s_wait_asynccnt 0" ::: "memory"); }

// ---------------------------------------------------------------------------
// K0: fp32 (R x C) -> bf16 transposed (C x R), batched over blockIdx.z
// ---------------------------------------------------------------------------
__global__ __launch_bounds__(256)
void transpose_to_bf16(const float* __restrict__ src, __bf16* __restrict__ dst,
                       int R, int C) {
  __shared__ float tile[32][33];
  size_t bo = (size_t)blockIdx.z * R * C;
  src += bo; dst += bo;
  int c0 = blockIdx.x * 32, r0 = blockIdx.y * 32;
  int tx = threadIdx.x & 31, ty = threadIdx.x >> 5;   // 32 x 8
  #pragma unroll
  for (int i = 0; i < 32; i += 8) {
    int r = r0 + ty + i, c = c0 + tx;
    if (r < R && c < C) tile[ty + i][tx] = src[(size_t)r * C + c];
  }
  __syncthreads();
  #pragma unroll
  for (int i = 0; i < 32; i += 8) {
    int c = c0 + ty + i, r = r0 + tx;
    if (r < R && c < C) dst[(size_t)c * R + r] = f2bf(tile[tx][ty + i]);
  }
}

// ---------------------------------------------------------------------------
// K1: row RMSNorm fp32 -> bf16. One block per token, 256 threads x 4 cols.
// ---------------------------------------------------------------------------
__global__ __launch_bounds__(256)
void rmsnorm_bf16(const float* __restrict__ x, const float* __restrict__ w,
                  __bf16* __restrict__ out) {
  __shared__ float red[256];
  int n = blockIdx.x, tid = threadIdx.x;
  const float* row = x + (size_t)n * D;
  float v[4]; float ss = 0.0f;
  #pragma unroll
  for (int i = 0; i < 4; ++i) { v[i] = row[tid * 4 + i]; ss += v[i] * v[i]; }
  red[tid] = ss; __syncthreads();
  for (int s = 128; s > 0; s >>= 1) {
    if (tid < s) red[tid] += red[tid + s];
    __syncthreads();
  }
  float rstd = rsqrtf(red[0] * (1.0f / D) + EPS);
  #pragma unroll
  for (int i = 0; i < 4; ++i) {
    int c = tid * 4 + i;
    out[(size_t)n * D + c] = f2bf(v[i] * rstd * w[c]);
  }
}

// ---------------------------------------------------------------------------
// K2: WMMA GEMM.  C[M,N] = A[M,K](bf16 row-major) * Bt[N,K]^T (bf16).
// 256 threads = 8 waves; block tile 128x128, BK=32; each wave owns a 32x64
// strip (2 A-frags x 4 B-frags = 8 WMMA per K-step). Tiles staged with
// async GLOBAL->LDS (b128), ASYNCcnt-waited before the barrier.
// mode 0: plain rows.  mode 1: gather A rows via slot_token (expert z),
// silu->bf16 out at slot rows.  mode 2: contiguous slot rows (expert z).
// act 0: f32 out.  act 1: silu -> bf16 out.
// ---------------------------------------------------------------------------
__global__ __launch_bounds__(256)
void gemm_bf16_wmma(const __bf16* __restrict__ A, const __bf16* __restrict__ Bt,
                    float* __restrict__ Cf, __bf16* __restrict__ Cb,
                    int M, int Nn, int K, int act, int mode,
                    const int* __restrict__ slot_token,
                    const int* __restrict__ counts,
                    const int* __restrict__ offsets) {
  int e = blockIdx.z;
  int off = 0, cnt = M;
  if (mode != 0) {
    off = offsets[e]; cnt = counts[e];
    Bt += (size_t)e * Nn * K;
  }
  int m0 = blockIdx.y * 128;
  if (mode != 0 && m0 >= cnt) return;
  int n0 = blockIdx.x * 128;

  __shared__ __bf16 As[128 * 40];   // 128 rows x 32 K-halves, pitch 40
  __shared__ __bf16 Bs[128 * 40];

  int tid = threadIdx.x;
  int wid = tid >> 5, lane = tid & 31;
  int g = lane >> 4, ml = lane & 15;
  int wm = wid >> 1, wn = wid & 1;  // wave grid 4 (M) x 2 (N)
  int lr = tid >> 1;                // tile row this thread stages (0..127)
  int lseg = (tid & 1) * 16;        // half-row segment (halves)

  // Resolve this thread's A source row once (constant over the K loop)
  int rm = m0 + lr;
  bool a_valid; const __bf16* arow = nullptr;
  if (mode == 0) {
    a_valid = rm < M;
    if (a_valid) arow = A + (size_t)rm * K;
  } else {
    a_valid = rm < cnt;
    if (a_valid) {
      int slot = off + rm;
      int ar = (mode == 1) ? slot_token[slot] : slot;
      arow = A + (size_t)ar * K;
    }
  }
  int rn = n0 + lr;
  bool b_valid = rn < Nn;
  const __bf16* brow = b_valid ? (Bt + (size_t)rn * K) : nullptr;

  unsigned as_dst = lds_byte_addr(&As[lr * 40 + lseg]);
  unsigned bs_dst = lds_byte_addr(&Bs[lr * 40 + lseg]);

  // invalid rows: zero-fill once (async path never touches them)
  if (!a_valid) *(v16bf*)&As[lr * 40 + lseg] = zero16bf();
  if (!b_valid) *(v16bf*)&Bs[lr * 40 + lseg] = zero16bf();

  v8f acc[2][4];
  #pragma unroll
  for (int i = 0; i < 2; ++i)
    #pragma unroll
    for (int j = 0; j < 4; ++j) acc[i][j] = zero8f();

  int ktiles = K >> 5;
  for (int kt = 0; kt < ktiles; ++kt) {
    int k0 = kt << 5;
    if (a_valid) {
      const void* src = arow + k0 + lseg;
      async_b128(as_dst, src);
      async_b128_o16(as_dst, src);
    }
    if (b_valid) {
      const void* src = brow + k0 + lseg;
      async_b128(bs_dst, src);
      async_b128_o16(bs_dst, src);
    }
    wait_async0();
    __syncthreads();

    // A fragments: lane group g holds K halves [16g, 16g+16) contiguous
    v16bf a[2];
    #pragma unroll
    for (int mf = 0; mf < 2; ++mf) {
      const __bf16* pa = &As[(wm * 32 + mf * 16 + ml) * 40];
      v8bf alo = *(const v8bf*)(pa + 16 * g);
      v8bf ahi = *(const v8bf*)(pa + 16 * g + 8);
      #pragma unroll
      for (int i = 0; i < 8; ++i) { a[mf][i] = alo[i]; a[mf][8 + i] = ahi[i]; }
    }
    #pragma unroll
    for (int nt = 0; nt < 4; ++nt) {
      // B fragment: lane n=ml, K halves [16g, 16g+16) contiguous
      v16bf b = *(const v16bf*)&Bs[(wn * 64 + nt * 16 + ml) * 40 + 16 * g];
      #pragma unroll
      for (int mf = 0; mf < 2; ++mf)
        acc[mf][nt] = wmma_bf16(a[mf], b, acc[mf][nt]);
    }
    __syncthreads();
  }

  // epilogue: C layout M = 8g + r, N = ml (+16*nt)
  #pragma unroll
  for (int mf = 0; mf < 2; ++mf) {
    #pragma unroll
    for (int nt = 0; nt < 4; ++nt) {
      #pragma unroll
      for (int r = 0; r < 8; ++r) {
        int mrow = wm * 32 + mf * 16 + 8 * g + r;
        int rme = m0 + mrow;
        int col = n0 + wn * 64 + nt * 16 + ml;
        bool valid = (mode == 0) ? (rme < M) : (rme < cnt);
        if (!valid || col >= Nn) continue;
        size_t crow = (mode == 0) ? (size_t)rme : (size_t)(off + rme);
        float v = acc[mf][nt][r];
        if (act == 1) {
          float sv = v / (1.0f + expf(-v));
          Cb[crow * Nn + col] = f2bf(sv);
        } else {
          Cf[crow * Nn + col] = v;
        }
      }
    }
  }
}

// ---------------------------------------------------------------------------
// K3: RoPE + repack. qkv f32 (N, 1536) -> Q bf16 (B,H,T,HD) scaled by
// HD^-0.5*log2e (so attention softmax runs in exp2 domain), K bf16
// (B,KVH,T,HD), V transposed bf16 (B,KVH,HD,T) for contiguous PV B-frags.
// ---------------------------------------------------------------------------
__global__ __launch_bounds__(256)
void rope_pack(const float* __restrict__ qkv, __bf16* __restrict__ Q,
               __bf16* __restrict__ Kk, __bf16* __restrict__ Vt) {
  int n = blockIdx.x; int b = n / T; int t = n % T;
  int tid = threadIdx.x;
  const float* row = qkv + (size_t)n * QKV;
  const float qscale = 0.125f * LOG2E;   // HD^-0.5 * log2(e)
  // q: 512 pairs
  for (int p = tid; p < 512; p += 256) {
    int hq = p >> 5, j = p & 31;
    float fr = powf(10000.0f, -(float)j * (1.0f / 32.0f));
    float ang = (float)t * fr;
    float cs = cosf(ang), sn = sinf(ang);
    float x0 = row[hq * 64 + 2 * j], x1 = row[hq * 64 + 2 * j + 1];
    __bf16* dst = Q + ((size_t)(b * H + hq) * T + t) * HD;
    dst[2 * j]     = f2bf((x0 * cs - x1 * sn) * qscale);
    dst[2 * j + 1] = f2bf((x0 * sn + x1 * cs) * qscale);
  }
  // k: 128 pairs
  if (tid < 128) {
    int kh = tid >> 5, j = tid & 31;
    float fr = powf(10000.0f, -(float)j * (1.0f / 32.0f));
    float ang = (float)t * fr;
    float cs = cosf(ang), sn = sinf(ang);
    float x0 = row[D + kh * 64 + 2 * j], x1 = row[D + kh * 64 + 2 * j + 1];
    __bf16* dst = Kk + ((size_t)(b * KVH + kh) * T + t) * HD;
    dst[2 * j]     = f2bf(x0 * cs - x1 * sn);
    dst[2 * j + 1] = f2bf(x0 * sn + x1 * cs);
  }
  // v: 256 elems, transposed store
  {
    int kh = tid >> 6, hd = tid & 63;
    float v = row[D + KVH * HD + tid];
    Vt[((size_t)(b * KVH + kh) * HD + hd) * T + t] = f2bf(v);
  }
}

// ---------------------------------------------------------------------------
// K4: flash attention. Grid (T/64, B*H), 128 threads = 4 waves; each wave
// owns 16 q rows, iterates causal kv in 32-key blocks, all matmuls WMMA.
// ---------------------------------------------------------------------------
__global__ __launch_bounds__(128)
void flash_attn(const __bf16* __restrict__ Q, const __bf16* __restrict__ Kk,
                const __bf16* __restrict__ Vt, __bf16* __restrict__ O) {
  __shared__ __bf16 pbuf[4][16][32];   // per-wave P staging (C-layout -> A-frag)
  int bh = blockIdx.y;
  int b = bh / H, h = bh % H, kh = h / (H / KVH);
  int tid = threadIdx.x, wid = tid >> 5, lane = tid & 31;
  int g = lane >> 4, ml = lane & 15;
  int q0 = blockIdx.x * 64 + wid * 16;

  const __bf16* Qb = Q  + ((size_t)(b * H + h) * T) * HD;
  const __bf16* Kb = Kk + ((size_t)(b * KVH + kh) * T) * HD;
  const __bf16* Vb = Vt + ((size_t)(b * KVH + kh) * HD) * T;

  // Q fragments (16 x 64 = two A-frags over HD)
  v16bf aq0, aq1;
  {
    const __bf16* qr = Qb + (size_t)(q0 + ml) * HD;
    v8bf x0 = *(const v8bf*)(qr + 16 * g);
    v8bf x1 = *(const v8bf*)(qr + 16 * g + 8);
    v8bf x2 = *(const v8bf*)(qr + 32 + 16 * g);
    v8bf x3 = *(const v8bf*)(qr + 40 + 16 * g);
    #pragma unroll
    for (int i = 0; i < 8; ++i) {
      aq0[i] = x0[i]; aq0[8 + i] = x1[i];
      aq1[i] = x2[i]; aq1[8 + i] = x3[i];
    }
  }

  v8f o[4];
  #pragma unroll
  for (int i = 0; i < 4; ++i) o[i] = zero8f();
  float mrow[8], lrow[8];
  #pragma unroll
  for (int r = 0; r < 8; ++r) { mrow[r] = -1e30f; lrow[r] = 0.0f; }

  int kend = q0 + 16;
  for (int kb = 0; kb < kend; kb += 32) {
    // S = Q K^T  (two 16x16 tiles over 32 keys)
    v8f c0 = zero8f(), c1 = zero8f();
    {
      const __bf16* kr = Kb + (size_t)(kb + ml) * HD;
      v16bf b0 = *(const v16bf*)(kr + 16 * g);
      v16bf b1 = *(const v16bf*)(kr + 32 + 16 * g);
      c0 = wmma_bf16(aq0, b0, c0);
      c0 = wmma_bf16(aq1, b1, c0);
      const __bf16* kr2 = kr + (size_t)16 * HD;
      v16bf b2 = *(const v16bf*)(kr2 + 16 * g);
      v16bf b3 = *(const v16bf*)(kr2 + 32 + 16 * g);
      c1 = wmma_bf16(aq0, b2, c1);
      c1 = wmma_bf16(aq1, b3, c1);
    }
    // causal mask + online softmax (log2 domain)
    #pragma unroll
    for (int r = 0; r < 8; ++r) {
      int rowq = q0 + 8 * g + r;
      if (kb + ml      > rowq) c0[r] = -1e30f;
      if (kb + 16 + ml > rowq) c1[r] = -1e30f;
    }
    float alpha[8];
    #pragma unroll
    for (int r = 0; r < 8; ++r) {
      float mb = fmaxf(c0[r], c1[r]);
      mb = fmaxf(mb, __shfl_xor(mb, 1));
      mb = fmaxf(mb, __shfl_xor(mb, 2));
      mb = fmaxf(mb, __shfl_xor(mb, 4));
      mb = fmaxf(mb, __shfl_xor(mb, 8));
      float mnew = fmaxf(mrow[r], mb);
      alpha[r] = exp2f(mrow[r] - mnew);
      float p0 = exp2f(c0[r] - mnew);
      float p1 = exp2f(c1[r] - mnew);
      float rs = p0 + p1;
      rs += __shfl_xor(rs, 1);
      rs += __shfl_xor(rs, 2);
      rs += __shfl_xor(rs, 4);
      rs += __shfl_xor(rs, 8);
      lrow[r] = lrow[r] * alpha[r] + rs;
      mrow[r] = mnew;
      pbuf[wid][8 * g + r][ml]      = f2bf(p0);
      pbuf[wid][8 * g + r][16 + ml] = f2bf(p1);
      o[0][r] *= alpha[r]; o[1][r] *= alpha[r];
      o[2][r] *= alpha[r]; o[3][r] *= alpha[r];
    }
    // wave-local LDS ordering (DS ops are in-order per wave; fence compiler)
    asm volatile("s_wait_dscnt 0" ::: "memory");
    v16bf ap;
    {
      const __bf16* pr = &pbuf[wid][ml][0];
      v8bf plo = *(const v8bf*)(pr + 16 * g);
      v8bf phi = *(const v8bf*)(pr + 16 * g + 8);
      #pragma unroll
      for (int i = 0; i < 8; ++i) { ap[i] = plo[i]; ap[8 + i] = phi[i]; }
    }
    // O += P * V  (V transposed: B-frag rows contiguous)
    #pragma unroll
    for (int t4 = 0; t4 < 4; ++t4) {
      v16bf bv = *(const v16bf*)(Vb + (size_t)(t4 * 16 + ml) * T + kb + 16 * g);
      o[t4] = wmma_bf16(ap, bv, o[t4]);
    }
    asm volatile("" ::: "memory");  // keep next-iter pbuf stores after ap loads
  }

  #pragma unroll
  for (int t4 = 0; t4 < 4; ++t4) {
    #pragma unroll
    for (int r = 0; r < 8; ++r) {
      int row = q0 + 8 * g + r;
      float ov = o[t4][r] / lrow[r];
      O[(size_t)(b * T + row) * D + h * HD + t4 * 16 + ml] = f2bf(ov);
    }
  }
}

// ---------------------------------------------------------------------------
// K5: x1 = x + wo_out; h2 = rmsnorm(x1); gate scores, top-2, expert counts.
// ---------------------------------------------------------------------------
__global__ __launch_bounds__(256)
void x1_gate_kernel(const float* __restrict__ x, const float* __restrict__ wo_out,
                    const float* __restrict__ fw, const float* __restrict__ gw,
                    const float* __restrict__ gb, float* __restrict__ x1,
                    __bf16* __restrict__ h2, float* __restrict__ topkw,
                    int* __restrict__ topki, int* __restrict__ counts) {
  __shared__ float red[256];
  __shared__ float gred[8][8];
  __shared__ float sc8[8];
  int n = blockIdx.x, tid = threadIdx.x;
  size_t base = (size_t)n * D;
  float v[4]; float ss = 0.0f;
  #pragma unroll
  for (int i = 0; i < 4; ++i) {
    int c = tid * 4 + i;
    v[i] = x[base + c] + wo_out[base + c];
    x1[base + c] = v[i];
    ss += v[i] * v[i];
  }
  red[tid] = ss; __syncthreads();
  for (int s = 128; s > 0; s >>= 1) {
    if (tid < s) red[tid] += red[tid + s];
    __syncthreads();
  }
  float rstd = rsqrtf(red[0] * (1.0f / D) + EPS);
  float hh[4];
  float pe[8];
  #pragma unroll
  for (int e2 = 0; e2 < 8; ++e2) pe[e2] = 0.0f;
  #pragma unroll
  for (int i = 0; i < 4; ++i) {
    int c = tid * 4 + i;
    hh[i] = v[i] * rstd * fw[c];
    h2[base + c] = f2bf(hh[i]);
    #pragma unroll
    for (int e2 = 0; e2 < 8; ++e2) pe[e2] += hh[i] * gw[(size_t)c * E + e2];
  }
  int wid = tid >> 5, lane = tid & 31;
  #pragma unroll
  for (int e2 = 0; e2 < 8; ++e2) {
    float s = pe[e2];
    s += __shfl_xor(s, 16); s += __shfl_xor(s, 8);
    s += __shfl_xor(s, 4);  s += __shfl_xor(s, 2);
    s += __shfl_xor(s, 1);
    if (lane == 0) gred[wid][e2] = s;
  }
  __syncthreads();
  if (tid < 8) {
    float s = 0.0f;
    #pragma unroll
    for (int w = 0; w < 8; ++w) s += gred[w][tid];
    sc8[tid] = 1.0f / (1.0f + expf(-s)) + gb[tid];
  }
  __syncthreads();
  if (tid == 0) {
    int i0 = 0; float b0 = sc8[0];
    for (int e2 = 1; e2 < 8; ++e2) if (sc8[e2] > b0) { b0 = sc8[e2]; i0 = e2; }
    int i1 = (i0 == 0) ? 1 : 0; float b1 = sc8[i1];
    for (int e2 = 0; e2 < 8; ++e2)
      if (e2 != i0 && sc8[e2] > b1) { b1 = sc8[e2]; i1 = e2; }
    float sum = b0 + b1 + 1e-20f;
    topkw[2 * n]     = b0 / sum;
    topkw[2 * n + 1] = b1 / sum;
    topki[2 * n]     = i0;
    topki[2 * n + 1] = i1;
    atomicAdd(&counts[i0], 1);
    atomicAdd(&counts[i1], 1);
  }
}

__global__ void init_counts(int* counts) {
  if (threadIdx.x < E && blockIdx.x == 0) counts[threadIdx.x] = 0;
}

__global__ void scan_offsets(const int* counts, int* offsets, int* cursor) {
  if (threadIdx.x == 0 && blockIdx.x == 0) {
    int acc = 0;
    for (int e = 0; e < E; ++e) { offsets[e] = acc; acc += counts[e]; cursor[e] = 0; }
  }
}

__global__ __launch_bounds__(256)
void gather_slots(const int* __restrict__ topki, const int* __restrict__ offsets,
                  int* __restrict__ cursor, int* __restrict__ slot_token,
                  int* __restrict__ slot_of) {
  int n = blockIdx.x * 256 + threadIdx.x;
  if (n >= N) return;
  #pragma unroll
  for (int k = 0; k < TOPK; ++k) {
    int e = topki[2 * n + k];
    int pos = atomicAdd(&cursor[e], 1);
    int slot = offsets[e] + pos;
    slot_token[slot] = n;
    slot_of[2 * n + k] = slot;
  }
}

// ---------------------------------------------------------------------------
// K8: out = x1 + shared + w0*routed[slot0] + w1*routed[slot1]
// ---------------------------------------------------------------------------
__global__ __launch_bounds__(256)
void final_combine(const float* __restrict__ x1, const float* __restrict__ sh,
                   const float* __restrict__ rout, const float* __restrict__ topkw,
                   const int* __restrict__ slot_of, float* __restrict__ out) {
  int gid = blockIdx.x * 256 + threadIdx.x;
  int base = gid * 4;
  int n = base / D, d = base % D;
  int s0 = slot_of[2 * n], s1 = slot_of[2 * n + 1];
  float w0 = topkw[2 * n], w1 = topkw[2 * n + 1];
  #pragma unroll
  for (int i = 0; i < 4; ++i) {
    out[base + i] = x1[base + i] + sh[base + i] +
                    w0 * rout[(size_t)s0 * D + d + i] +
                    w1 * rout[(size_t)s1 * D + d + i];
  }
}

// ---------------------------------------------------------------------------
extern "C" void kernel_launch(void* const* d_in, const int* in_sizes, int n_in,
                              void* d_out, int out_size, void* d_ws, size_t ws_size,
                              hipStream_t stream) {
  (void)in_sizes; (void)n_in; (void)out_size; (void)ws_size;
  const float* x          = (const float*)d_in[0];
  const float* attn_norm  = (const float*)d_in[1];
  const float* wq         = (const float*)d_in[2];
  const float* wk         = (const float*)d_in[3];
  const float* wv         = (const float*)d_in[4];
  const float* wo         = (const float*)d_in[5];
  const float* ffn_norm   = (const float*)d_in[6];
  const float* gate_w     = (const float*)d_in[7];
  const float* gate_bias  = (const float*)d_in[8];
  const float* ew1        = (const float*)d_in[9];
  const float* ew2        = (const float*)d_in[10];
  const float* sw1        = (const float*)d_in[11];
  const float* sw2        = (const float*)d_in[12];
  float* out = (float*)d_out;

  char* p = (char*)d_ws;
  auto alloc = [&](size_t bytes) -> void* {
    void* r = (void*)p; p += (bytes + 255) & ~(size_t)255; return r;
  };
  __bf16* wqkvT  = (__bf16*)alloc((size_t)QKV * D * 2);
  __bf16* woT    = (__bf16*)alloc((size_t)D * D * 2);
  __bf16* sw1T   = (__bf16*)alloc((size_t)FFH * D * 2);
  __bf16* sw2T   = (__bf16*)alloc((size_t)D * FFH * 2);
  __bf16* ew1T   = (__bf16*)alloc((size_t)E * FFH * D * 2);
  __bf16* ew2T   = (__bf16*)alloc((size_t)E * D * FFH * 2);
  __bf16* h_bf   = (__bf16*)alloc((size_t)N * D * 2);
  float*  qkv_f  = (float*)alloc((size_t)N * QKV * 4);
  __bf16* q_bf   = (__bf16*)alloc((size_t)N * D * 2);
  __bf16* k_bf   = (__bf16*)alloc((size_t)Bb * KVH * T * HD * 2);
  __bf16* vT_bf  = (__bf16*)alloc((size_t)Bb * KVH * HD * T * 2);
  __bf16* attn_bf= (__bf16*)alloc((size_t)N * D * 2);
  float*  wo_out = (float*)alloc((size_t)N * D * 4);
  float*  x1     = (float*)alloc((size_t)N * D * 4);
  __bf16* h2_bf  = (__bf16*)alloc((size_t)N * D * 2);
  __bf16* smid   = (__bf16*)alloc((size_t)N * FFH * 2);
  float*  sh_out = (float*)alloc((size_t)N * D * 4);
  __bf16* rmid   = (__bf16*)alloc((size_t)N * TOPK * FFH * 2);
  float*  r_out  = (float*)alloc((size_t)N * TOPK * D * 4);
  float*  topkw  = (float*)alloc((size_t)N * TOPK * 4);
  int*    topki  = (int*)alloc((size_t)N * TOPK * 4);
  int*    counts = (int*)alloc(E * 4);
  int*    offs   = (int*)alloc(E * 4);
  int*    cursor = (int*)alloc(E * 4);
  int*    slot_token = (int*)alloc((size_t)N * TOPK * 4);
  int*    slot_of    = (int*)alloc((size_t)N * TOPK * 4);

  dim3 b256(256), b128(128);

  init_counts<<<1, 32, 0, stream>>>(counts);

  // weight transposes (fp32 -> bf16, K-major for WMMA B operands)
  transpose_to_bf16<<<dim3(D / 32, D / 32, 1), b256, 0, stream>>>(wq, wqkvT, D, D);
  transpose_to_bf16<<<dim3((KVH * HD) / 32, D / 32, 1), b256, 0, stream>>>(
      wk, wqkvT + (size_t)D * D, D, KVH * HD);
  transpose_to_bf16<<<dim3((KVH * HD) / 32, D / 32, 1), b256, 0, stream>>>(
      wv, wqkvT + (size_t)(D + KVH * HD) * D, D, KVH * HD);
  transpose_to_bf16<<<dim3(D / 32, D / 32, 1), b256, 0, stream>>>(wo, woT, D, D);
  transpose_to_bf16<<<dim3(FFH / 32, D / 32, 1), b256, 0, stream>>>(sw1, sw1T, D, FFH);
  transpose_to_bf16<<<dim3(D / 32, FFH / 32, 1), b256, 0, stream>>>(sw2, sw2T, FFH, D);
  transpose_to_bf16<<<dim3(FFH / 32, D / 32, E), b256, 0, stream>>>(ew1, ew1T, D, FFH);
  transpose_to_bf16<<<dim3(D / 32, FFH / 32, E), b256, 0, stream>>>(ew2, ew2T, FFH, D);

  // attention path
  rmsnorm_bf16<<<N, b256, 0, stream>>>(x, attn_norm, h_bf);
  gemm_bf16_wmma<<<dim3(QKV / 128, N / 128, 1), b256, 0, stream>>>(
      h_bf, wqkvT, qkv_f, nullptr, N, QKV, D, 0, 0, nullptr, nullptr, nullptr);
  rope_pack<<<N, b256, 0, stream>>>(qkv_f, q_bf, k_bf, vT_bf);
  flash_attn<<<dim3(T / 64, Bb * H), b128, 0, stream>>>(q_bf, k_bf, vT_bf, attn_bf);
  gemm_bf16_wmma<<<dim3(D / 128, N / 128, 1), b256, 0, stream>>>(
      attn_bf, woT, wo_out, nullptr, N, D, D, 0, 0, nullptr, nullptr, nullptr);

  // gate + routing
  x1_gate_kernel<<<N, b256, 0, stream>>>(x, wo_out, ffn_norm, gate_w, gate_bias,
                                         x1, h2_bf, topkw, topki, counts);
  scan_offsets<<<1, 1, 0, stream>>>(counts, offs, cursor);
  gather_slots<<<(N + 255) / 256, b256, 0, stream>>>(topki, offs, cursor,
                                                     slot_token, slot_of);

  // shared expert
  gemm_bf16_wmma<<<dim3(FFH / 128, N / 128, 1), b256, 0, stream>>>(
      h2_bf, sw1T, nullptr, smid, N, FFH, D, 1, 0, nullptr, nullptr, nullptr);
  gemm_bf16_wmma<<<dim3(D / 128, N / 128, 1), b256, 0, stream>>>(
      smid, sw2T, sh_out, nullptr, N, D, FFH, 0, 0, nullptr, nullptr, nullptr);

  // routed experts (gathered, top-2 only)
  gemm_bf16_wmma<<<dim3(FFH / 128, N / 128, E), b256, 0, stream>>>(
      h2_bf, ew1T, nullptr, rmid, N, FFH, D, 1, 1, slot_token, counts, offs);
  gemm_bf16_wmma<<<dim3(D / 128, N / 128, E), b256, 0, stream>>>(
      rmid, ew2T, r_out, nullptr, N, D, FFH, 0, 2, slot_token, counts, offs);

  final_combine<<<(N * D) / 1024, b256, 0, stream>>>(x1, sh_out, r_out, topkw,
                                                     slot_of, out);
}
